// Fedformer_21131239096337
// MI455X (gfx1250) — matrix operand order repaired and checked by reading, not compiled, not run on hardware
//
#include <hip/hip_runtime.h>
#include <math.h>

// ---------------------------------------------------------------------------
// FEDformer FourierCrossAttention on MI455X (gfx1250, wave32, WMMA).
// Truncated rfft/irfft -> dense DFT/iDFT GEMMs on v_wmma_f32_16x16x32_f16.
// All intermediates stored pre-swizzled in WMMA fragment layout so operand
// loads are 2x b128 per fragment (no u16 gathers, no half packing).
// B=32, L=1024, H=8, E=64, MODES=64.
// ---------------------------------------------------------------------------

typedef __attribute__((ext_vector_type(16))) _Float16 v16h;
typedef __attribute__((ext_vector_type(8)))  _Float16 v8h;
typedef __attribute__((ext_vector_type(8)))  float    v8f;

#define Bn    32
#define Hn    8
#define En    64
#define Ln    1024
#define Mn    64
#define BHn   256

__device__ __forceinline__ v8f wmma16(v16h a, v16h b, v8f c) {
    return __builtin_amdgcn_wmma_f32_16x16x32_f16(
        false, a, false, b, (short)0, c, false, false);
}

// ---- fragment-layout offset helpers (16-bit operands, contiguous k order) ---------
// A fragment of a 64(M) x 64(K) matrix: element (M,K) -> flat offset.
// Fragment grid: rm = M>>4 (4 row tiles), kc = K>>5 (2 k-chunks of 32).
// Within a chunk: lane = h32*16 + (M&15); K = i + 16*h32.
__device__ __forceinline__ int aoff64(int M, int K) {
    int kl  = K & 31;
    int h32 = kl >> 4;
    int i   = kl & 15;
    int lane = h32 * 16 + (M & 15);
    return (((M >> 4) * 2 + (K >> 5)) * 32 + lane) * 16 + i;
}
// B fragment of a 64(K) x 64(N) matrix: element (K,N) -> flat offset.
// Within a chunk: lane = h32*16 + (N&15); K = i + 16*h32.
__device__ __forceinline__ int boff64(int K, int N) {
    int kl  = K & 31;
    int h32 = kl >> 4;
    int i   = kl & 15;
    int lane = h32 * 16 + (N & 15);
    return (((N >> 4) * 2 + (K >> 5)) * 32 + lane) * 16 + i;
}
// A-fragment layout for the per-(h,mode) weight GEMM: M=b (32), K=e (64).
__device__ __forceinline__ size_t woff(int h, int mode, int b, int e) {
    int el  = e & 31;
    int h32 = el >> 4;
    int i   = el & 15;
    int lane = h32 * 16 + (b & 15);
    return ((((size_t)(h * 64 + mode) * 2 + (b >> 4)) * 2 + (e >> 5)) * 32 + lane) * 16 + i;
}

// complex tanh: tanh(a+ib) = (sinh 2a + i sin 2b) / (cosh 2a + cos 2b)
__device__ __forceinline__ void ctanhf(float a, float b, float& re, float& im) {
    float a2 = 2.0f * a, b2 = 2.0f * b;
    if (fabsf(a2) > 20.0f) { re = (a2 > 0.0f) ? 1.0f : -1.0f; im = 0.0f; return; }
    float sb, cb; __sincosf(b2, &sb, &cb);
    float ea = __expf(a2), ia = 1.0f / ea;
    float d  = 0.5f * (ea + ia) + cb;
    re = 0.5f * (ea - ia) / d;
    im = sb / d;
}

// ---------------------------------------------------------------------------
// Kernel 0: bases, generated directly in B-fragment order.
// Fr/Fi: forward DFT e^{-i 2 pi m t / L}; fragment grid (cm:4, kc:32).
// Gsw:   iDFT basis rows k=2m (c_m cos) / k=2m+1 (-c_m sin), c_0=1, c_m=2;
//        fragment grid (tn:64 column tiles, kc:4).
// ---------------------------------------------------------------------------
__global__ void k_basis(_Float16* __restrict__ Fr, _Float16* __restrict__ Fi,
                        _Float16* __restrict__ Gsw) {
    const float STEP = 6.283185307179586f / 1024.0f;
    int stride = gridDim.x * blockDim.x;
    for (int idx = blockIdx.x * blockDim.x + threadIdx.x; idx < 65536; idx += stride) {
        int i = idx & 15, lane = (idx >> 4) & 31, kc = (idx >> 9) & 31, cm = idx >> 14;
        int m = cm * 16 + (lane & 15);
        int t = kc * 32 + i + ((lane & 16) ? 16 : 0);
        int r = (m * t) & 1023;
        float s, c; __sincosf((float)r * STEP, &s, &c);
        Fr[idx] = (_Float16)c;
        Fi[idx] = (_Float16)(-s);
    }
    for (int idx = blockIdx.x * blockDim.x + threadIdx.x; idx < 131072; idx += stride) {
        int i = idx & 15, lane = (idx >> 4) & 31, kc = (idx >> 9) & 3, tn = idx >> 11;
        int t = tn * 16 + (lane & 15);
        int kk = kc * 32 + i + ((lane & 16) ? 16 : 0);
        int m = kk >> 1;
        int r = (m * t) & 1023;
        float s, c; __sincosf((float)r * STEP, &s, &c);
        float coef = (m == 0) ? 1.0f : 2.0f;
        float v = (kk & 1) ? (-coef * s) : (coef * c);
        if (kk == 1) v = 0.0f;          // imag of mode 0 discarded by irfft
        Gsw[idx] = (_Float16)v;
    }
}

// ---------------------------------------------------------------------------
// Kernel 1: truncated DFT.  Per (b,h): [64 x 1024] fp32 x [1024 x 64] basis.
// A operand staged via LDS: coalesced b32 global loads -> f16 -> one
// ds_store_b128 per thread into [e][perm(t)] so each lane's A fragment is
// 16 contiguous halves (2x ds_load_b128).  B fragments: v16h from Fr/Fi.
// Output scattered u16 into the consumers' fragment layouts.
// ---------------------------------------------------------------------------
__global__ void k_dft(const float* __restrict__ x,
                      const _Float16* __restrict__ Fr, const _Float16* __restrict__ Fi,
                      _Float16* __restrict__ OAr, _Float16* __restrict__ OAi,
                      _Float16* __restrict__ OBr, _Float16* __restrict__ OBi,
                      int swapMK)   // 0: A-role (M=mode,K=e) [q]; 1: (M=e,K=mode) [k]
{
    __shared__ _Float16 lds[En * 32];           // [e][perm(t_local)] f16, 4 KB
    int bh = blockIdx.x, b = bh >> 3, h = bh & 7;
    int tid = threadIdx.x;
    int wave = tid >> 5, lane = tid & 31;
    int rm = wave >> 1, cm0 = (wave & 1) * 2;
    int e_lane = rm * 16 + (lane & 15);
    // staging roles: thread -> (e = tid&63, group of 8 t)
    int se = tid & 63;
    int sg = tid >> 6;                          // 0..3
    int pstart = sg * 8;                              // {0,8,16,24}: 8-blocks in t order
    const float* xb = x + ((size_t)b * Ln * Hn + h) * En;

    v8f cr[2] = {{}, {}}, ci[2] = {{}, {}};
    for (int kc = 0; kc < 32; ++kc) {
        __syncthreads();
        {
            const float* src = xb + (size_t)(kc * 32 + sg * 8) * (Hn * En) + se;
            v8h vals;
#pragma unroll
            for (int j = 0; j < 8; ++j)
                vals[j] = (_Float16)src[(size_t)j * (Hn * En)];
            *(v8h*)&lds[se * 32 + pstart] = vals;
        }
        __syncthreads();
        v16h a;
        {
            v8h lo = *(const v8h*)&lds[e_lane * 32 + (lane & 16)];
            v8h hi = *(const v8h*)&lds[e_lane * 32 + (lane & 16) + 8];
            a = __builtin_shufflevector(lo, hi, 0,1,2,3,4,5,6,7,8,9,10,11,12,13,14,15);
        }
        size_t fo0 = ((size_t)((cm0 * 32 + kc) * 32 + lane)) * 16;
        size_t fo1 = ((size_t)(((cm0 + 1) * 32 + kc) * 32 + lane)) * 16;
        v16h br0 = *(const v16h*)(Fr + fo0), bi0 = *(const v16h*)(Fi + fo0);
        v16h br1 = *(const v16h*)(Fr + fo1), bi1 = *(const v16h*)(Fi + fo1);
        cr[0] = wmma16(a, br0, cr[0]);  ci[0] = wmma16(a, bi0, ci[0]);
        cr[1] = wmma16(a, br1, cr[1]);  ci[1] = wmma16(a, bi1, ci[1]);
    }
    size_t base = (size_t)bh * 4096;
#pragma unroll
    for (int r = 0; r < 8; ++r) {
        int ep = rm * 16 + r + ((lane & 16) ? 8 : 0);
#pragma unroll
        for (int cc = 0; cc < 2; ++cc) {
            int mp = (cm0 + cc) * 16 + (lane & 15);
            float vr = cr[cc][r], vi = ci[cc][r];
            int ao = swapMK ? aoff64(ep, mp) : aoff64(mp, ep);
            OAr[base + ao] = (_Float16)vr;
            OAi[base + ao] = (_Float16)vi;
            if (OBr != nullptr) {
                int bo = boff64(ep, mp);    // B-role: K-dim = e, N = mode
                OBr[base + bo] = (_Float16)vr;
                OBi[base + bo] = (_Float16)vi;
            }
        }
    }
}

// ---------------------------------------------------------------------------
// Kernel 2: S[x][y] = ctanh( sum_e Q[e][x] K[e][y] ), complex 64x64 per (b,h).
// Qa: A fragments (M=x, K=e); Kb: B fragments (K=e, N=y).  Output into the
// B-fragment layout of k_xqkv (K-dim=y, N=x).
// ---------------------------------------------------------------------------
__global__ void k_scores(const _Float16* __restrict__ Qa_r, const _Float16* __restrict__ Qa_i,
                         const _Float16* __restrict__ Kb_r, const _Float16* __restrict__ Kb_i,
                         _Float16* __restrict__ Sb_r, _Float16* __restrict__ Sb_i) {
    int bh = blockIdx.x;
    int wave = threadIdx.x >> 5, lane = threadIdx.x & 31;
    int rm = wave >> 1, cm0 = (wave & 1) * 2;
    size_t base = (size_t)bh * 4096;

    v8f cr[2] = {{}, {}}, ci[2] = {{}, {}};
    for (int kc = 0; kc < 2; ++kc) {
        size_t ao = base + ((size_t)((rm * 2 + kc) * 32 + lane)) * 16;
        v16h ar = *(const v16h*)(Qa_r + ao), ai = *(const v16h*)(Qa_i + ao);
        v16h an = -ai;
#pragma unroll
        for (int cc = 0; cc < 2; ++cc) {
            size_t bo = base + ((size_t)(((cm0 + cc) * 2 + kc) * 32 + lane)) * 16;
            v16h br = *(const v16h*)(Kb_r + bo), bi = *(const v16h*)(Kb_i + bo);
            cr[cc] = wmma16(ar, br, cr[cc]);  cr[cc] = wmma16(an, bi, cr[cc]);
            ci[cc] = wmma16(ar, bi, ci[cc]);  ci[cc] = wmma16(ai, br, ci[cc]);
        }
    }
#pragma unroll
    for (int r = 0; r < 8; ++r) {
        int xp = rm * 16 + r + ((lane & 16) ? 8 : 0);
#pragma unroll
        for (int cc = 0; cc < 2; ++cc) {
            int yp = (cm0 + cc) * 16 + (lane & 15);
            float tr, ti; ctanhf(cr[cc][r], ci[cc][r], tr, ti);
            int so = boff64(yp, xp);
            Sb_r[base + so] = (_Float16)tr;
            Sb_i[base + so] = (_Float16)ti;
        }
    }
}

// ---------------------------------------------------------------------------
// Kernel 3: xqkv[e][x] = sum_y K[e][y] S[x][y].  Ka: A frags (M=e, K=y);
// Sb: B frags (K=y, N=x).  Output into k_wmul's A layout (M=b, K=e per
// (h,mode)).
// ---------------------------------------------------------------------------
__global__ void k_xqkv(const _Float16* __restrict__ Ka_r, const _Float16* __restrict__ Ka_i,
                       const _Float16* __restrict__ Sb_r, const _Float16* __restrict__ Sb_i,
                       _Float16* __restrict__ Aw_r, _Float16* __restrict__ Aw_i) {
    int bh = blockIdx.x, b = bh >> 3, h = bh & 7;
    int wave = threadIdx.x >> 5, lane = threadIdx.x & 31;
    int rm = wave >> 1, cm0 = (wave & 1) * 2;
    size_t base = (size_t)bh * 4096;

    v8f cr[2] = {{}, {}}, ci[2] = {{}, {}};
    for (int kc = 0; kc < 2; ++kc) {
        size_t ao = base + ((size_t)((rm * 2 + kc) * 32 + lane)) * 16;
        v16h ar = *(const v16h*)(Ka_r + ao), ai = *(const v16h*)(Ka_i + ao);
        v16h an = -ai;
#pragma unroll
        for (int cc = 0; cc < 2; ++cc) {
            size_t bo = base + ((size_t)(((cm0 + cc) * 2 + kc) * 32 + lane)) * 16;
            v16h br = *(const v16h*)(Sb_r + bo), bi = *(const v16h*)(Sb_i + bo);
            cr[cc] = wmma16(ar, br, cr[cc]);  cr[cc] = wmma16(an, bi, cr[cc]);
            ci[cc] = wmma16(ar, bi, ci[cc]);  ci[cc] = wmma16(ai, br, ci[cc]);
        }
    }
#pragma unroll
    for (int r = 0; r < 8; ++r) {
        int ep = rm * 16 + r + ((lane & 16) ? 8 : 0);
#pragma unroll
        for (int cc = 0; cc < 2; ++cc) {
            int xp = (cm0 + cc) * 16 + (lane & 15);
            size_t wo = woff(h, xp, b, ep);
            Aw_r[wo] = (_Float16)cr[cc][r];
            Aw_i[wo] = (_Float16)ci[cc][r];
        }
    }
}

// ---------------------------------------------------------------------------
// Kernel 4: per-(h,mode): Out[b][o] = sum_e xqkv[b][e] W[e][o], complex.
// W pre-scaled by 2^18 (f16 range), compensated in the final store scale.
// Output packed into k_idft's A layout: per (b,h) a 64(o) x 128(k) matrix,
// k = 2*mode (Re) / 2*mode+1 (Im).
// ---------------------------------------------------------------------------
__global__ void k_wmul(const _Float16* __restrict__ Aw_r, const _Float16* __restrict__ Aw_i,
                       const float* __restrict__ wr, const float* __restrict__ wi,
                       _Float16* __restrict__ Psw) {
    int hx = blockIdx.x;                   // h*64 + mode
    int h = hx >> 6, mode = hx & 63;
    int wave = threadIdx.x >> 5, lane = threadIdx.x & 31;
    int rm = wave & 1, cm = wave >> 1;
    int o0 = cm * 16 + (lane & 15);

    v8f cre = {}, cim = {};
    for (int kc = 0; kc < 2; ++kc) {
        size_t ao = ((((size_t)(h * 64 + mode) * 2 + rm) * 2 + kc) * 32 + lane) * 16;
        v16h ar = *(const v16h*)(Aw_r + ao), ai = *(const v16h*)(Aw_i + ao);
        v16h an = -ai;
        v16h br, bi;
#pragma unroll
        for (int i = 0; i < 16; ++i) {     // W gather (16 MB read once, fp32)
            int e = kc * 32 + i + ((lane & 16) ? 16 : 0);
            size_t idx = ((size_t)(h * 64 + e) * 64 + o0) * 64 + mode;
            br[i] = (_Float16)(wr[idx] * 262144.0f);
            bi[i] = (_Float16)(wi[idx] * 262144.0f);
        }
        cre = wmma16(ar, br, cre);  cre = wmma16(an, bi, cre);
        cim = wmma16(ar, bi, cim);  cim = wmma16(ai, br, cim);
    }
#pragma unroll
    for (int r = 0; r < 8; ++r) {
        int bb = rm * 16 + r + ((lane & 16) ? 8 : 0);
        int bhp = bb * 8 + h;
#pragma unroll
        for (int c = 0; c < 2; ++c) {
            int kk = 2 * mode + c;
            int kcp = kk >> 5, kl = kk & 31;
            int h32 = kl >> 4;
            int ii  = kl & 15;
            int lp  = h32 * 16 + (o0 & 15);
            size_t off = (size_t)bhp * 8192 + ((size_t)((cm * 4 + kcp) * 32 + lp)) * 16 + ii;
            Psw[off] = (_Float16)(c ? cim[r] : cre[r]);
        }
    }
}

// ---------------------------------------------------------------------------
// Kernel 5: truncated iDFT.  Per (b,h): [64(o) x 128(k)] x [128 x 1024].
// Store scale = (1/1024) * 2^-18 * 2^-18 = 2^-46.
// ---------------------------------------------------------------------------
__global__ void k_idft(const _Float16* __restrict__ Psw, const _Float16* __restrict__ Gsw,
                       float* __restrict__ out) {
    int bh = blockIdx.x, tblk = blockIdx.y;
    int wave = threadIdx.x >> 5, lane = threadIdx.x & 31;
    int rm = wave >> 1, cbase = (wave & 1) * 4;
    size_t pb = (size_t)bh * 8192;

    v8f acc[4] = {{}, {}, {}, {}};
    for (int kc = 0; kc < 4; ++kc) {
        size_t ao = pb + ((size_t)((rm * 4 + kc) * 32 + lane)) * 16;
        v16h a = *(const v16h*)(Psw + ao);
#pragma unroll
        for (int ct = 0; ct < 4; ++ct) {
            int tn = tblk * 8 + cbase + ct;
            size_t go = ((size_t)((tn * 4 + kc) * 32 + lane)) * 16;
            v16h bfrag = *(const v16h*)(Gsw + go);
            acc[ct] = wmma16(a, bfrag, acc[ct]);
        }
    }
    const float SCALE = 1.4210854715202004e-14f;   // 2^-46
#pragma unroll
    for (int ct = 0; ct < 4; ++ct) {
        int t = (tblk * 8 + cbase + ct) * 16 + (lane & 15);
#pragma unroll
        for (int r = 0; r < 8; ++r) {
            int oo = rm * 16 + r + ((lane & 16) ? 8 : 0);
            out[((size_t)bh * 64 + oo) * 1024 + t] = acc[ct][r] * SCALE;
        }
    }
}

// ---------------------------------------------------------------------------
extern "C" void kernel_launch(void* const* d_in, const int* in_sizes, int n_in,
                              void* d_out, int out_size, void* d_ws, size_t ws_size,
                              hipStream_t stream) {
    const float* q  = (const float*)d_in[0];
    const float* k  = (const float*)d_in[1];
    // d_in[2] = v, unused by FourierCrossAttention's math
    const float* wr = (const float*)d_in[3];
    const float* wi = (const float*)d_in[4];

    // Workspace (all f16, ~24.5 MB), every buffer in its consumer's layout:
    _Float16* Fr   = (_Float16*)d_ws;              // 65536   DFT basis (B frags)
    _Float16* Fi   = Fr   + 65536;                 // 65536
    _Float16* Gsw  = Fi   + 65536;                 // 131072  iDFT basis (B frags)
    _Float16* Qa_r = Gsw  + 131072;                // 1 M each: Q spectrum, A frags
    _Float16* Qa_i = Qa_r + 1048576;
    _Float16* Kb_r = Qa_i + 1048576;               // K spectrum, B frags (scores)
    _Float16* Kb_i = Kb_r + 1048576;
    _Float16* Ka_r = Kb_i + 1048576;               // K spectrum, A frags (xqkv)
    _Float16* Ka_i = Ka_r + 1048576;
    _Float16* Sb_r = Ka_i + 1048576;               // tanh scores, B frags
    _Float16* Sb_i = Sb_r + 1048576;
    _Float16* Aw_r = Sb_i + 1048576;               // xqkv, A frags (wmul)
    _Float16* Aw_i = Aw_r + 1048576;
    _Float16* Psw  = Aw_i + 1048576;               // 2097152: packed, A frags (idft)

    k_basis <<<256, 256, 0, stream>>>(Fr, Fi, Gsw);
    k_dft   <<<BHn, 256, 0, stream>>>(q, Fr, Fi, Qa_r, Qa_i, nullptr, nullptr, 0);
    k_dft   <<<BHn, 256, 0, stream>>>(k, Fr, Fi, Ka_r, Ka_i, Kb_r, Kb_i, 1);
    k_scores<<<BHn, 256, 0, stream>>>(Qa_r, Qa_i, Kb_r, Kb_i, Sb_r, Sb_i);
    k_xqkv  <<<BHn, 256, 0, stream>>>(Ka_r, Ka_i, Sb_r, Sb_i, Aw_r, Aw_i);
    k_wmul  <<<Hn * Mn, 256, 0, stream>>>(Aw_r, Aw_i, wr, wi, Psw);
    dim3 g5(BHn, 8);
    k_idft  <<<g5, 256, 0, stream>>>(Psw, Gsw, (float*)d_out);
}
